// BoxMultiHeadedAttention_41575283425591
// MI455X (gfx1250) — matrix operand order repaired and checked by reading, not compiled, not run on hardware
//
#include <hip/hip_runtime.h>
#include <hip/hip_bf16.h>

// ---- problem constants (match reference) ----
#define Bc 8
#define Nn 512
#define Dd 512
#define Hh 8
#define Dk 64

typedef __attribute__((ext_vector_type(16))) __bf16 v16bf;
typedef __attribute__((ext_vector_type(8)))  __bf16 bf16x8;
typedef __attribute__((ext_vector_type(8)))  float  v8f;

__device__ __forceinline__ __bf16 to_bf16(float x)  { return (__bf16)x; }
__device__ __forceinline__ __bf16 to_bf16(__bf16 x) { return x; }

// Load 8 consecutive elements into fragment slots [lo, lo+8). 16B/32B aligned
// by construction (all k offsets are multiples of 8 elements).
__device__ __forceinline__ void load8(const __bf16* p, v16bf& f, int lo) {
    bf16x8 v = *(const bf16x8*)p;
    #pragma unroll
    for (int e = 0; e < 8; ++e) f[lo + e] = v[e];
}
__device__ __forceinline__ void load8(const float* p, v16bf& f, int lo) {
    float4 v0 = *(const float4*)p;
    float4 v1 = *(const float4*)(p + 4);
    f[lo + 0] = (__bf16)v0.x; f[lo + 1] = (__bf16)v0.y;
    f[lo + 2] = (__bf16)v0.z; f[lo + 3] = (__bf16)v0.w;
    f[lo + 4] = (__bf16)v1.x; f[lo + 5] = (__bf16)v1.y;
    f[lo + 6] = (__bf16)v1.z; f[lo + 7] = (__bf16)v1.w;
}

// ---------------------------------------------------------------------------
// Kernel 1: pairwise box embedding -> per-head relu linear -> log-clip bias.
// One thread per (b, n, m); writes wgLog[b,h,n,m] for all 8 heads.
// ---------------------------------------------------------------------------
__global__ __launch_bounds__(256)
void geo_bias_kernel(const float* __restrict__ box,
                     const float* __restrict__ WGw,
                     const float* __restrict__ WGb,
                     float* __restrict__ wgLog)
{
    __shared__ float sW[Hh * 64];
    __shared__ float sB[Hh];
    for (int i = threadIdx.x; i < Hh * 64; i += blockDim.x) sW[i] = WGw[i];
    if (threadIdx.x < Hh) sB[threadIdx.x] = WGb[threadIdx.x];
    __syncthreads();

    long t = (long)blockIdx.x * blockDim.x + threadIdx.x;     // over Bc*Nn*Nn
    int  m = (int)(t % Nn);
    long tt = t / Nn;
    int  n = (int)(tt % Nn);
    int  b = (int)(tt / Nn);

    const float4 bn = ((const float4*)box)[b * Nn + n];
    const float4 bm = ((const float4*)box)[b * Nn + m];

    float cxn = (bn.x + bn.z) * 0.5f, cyn = (bn.y + bn.w) * 0.5f;
    float wn  = bn.z - bn.x + 1.0f,   hn  = bn.w - bn.y + 1.0f;
    float cxm = (bm.x + bm.z) * 0.5f, cym = (bm.y + bm.w) * 0.5f;
    float wm  = bm.z - bm.x + 1.0f,   hm  = bm.w - bm.y + 1.0f;

    float pos[4];
    pos[0] = logf(fmaxf(fabsf((cxn - cxm) / wn), 1e-3f));
    pos[1] = logf(fmaxf(fabsf((cyn - cym) / hn), 1e-3f));
    pos[2] = logf(wn / wm);
    pos[3] = logf(hn / hm);

    float acc[Hh];
    #pragma unroll
    for (int h = 0; h < Hh; ++h) acc[h] = 0.f;

    const float LN1000 = 6.90775527898f;
    #pragma unroll
    for (int f = 0; f < 8; ++f) {
        float dm100 = 100.f * __expf(-(float)f * (0.125f * LN1000));
        #pragma unroll
        for (int p = 0; p < 4; ++p) {
            float ang = pos[p] * dm100;
            float s = __sinf(ang);
            float c = __cosf(ang);
            int g = p * 8 + f;
            #pragma unroll
            for (int h = 0; h < Hh; ++h)
                acc[h] += sW[h * 64 + g] * s + sW[h * 64 + 32 + g] * c;
        }
    }
    #pragma unroll
    for (int h = 0; h < Hh; ++h) {
        float wg = fmaxf(acc[h] + sB[h], 0.f);
        wgLog[(((long)(b * Hh + h) * Nn + n) * Nn) + m] = logf(fmaxf(wg, 1e-6f));
    }
}

// ---------------------------------------------------------------------------
// Kernel 2: batched WMMA GEMM. Each wave32 computes a 16 x (16*NT) strip:
// the A fragment is loaded once per k-step and reused across NT WMMAs.
//   C[m,n] = alpha * sum_k A[m,k] * (BT ? B[n,k] : B[k,n])
//            (+ rowBias[n] if HAS_BIAS) (+ addFull[idx] if HAS_ADD)
// All epilogue options are compile-time so no per-element branches/waits.
// ---------------------------------------------------------------------------
template <bool BT, typename TA, typename TB,
          bool HAS_BIAS, bool HAS_ADD, bool OUT_F, bool OUT_B, int NT>
__global__ void wmma_gemm_kernel(const TA* __restrict__ A, const TB* __restrict__ B,
                                 const float* __restrict__ rowBias,
                                 const float* __restrict__ addFull,
                                 float alpha,
                                 float* __restrict__ outF, __bf16* __restrict__ outB,
                                 int M, int Nc, int K,
                                 int lda, int ldb, int ldc,
                                 int Hdim,
                                 long sAb, long sAh, long sBb, long sBh,
                                 long sCb, long sCh)
{
    const int wpb  = blockDim.x >> 5;
    const int wave = threadIdx.x >> 5;
    const int lane = threadIdx.x & 31;
    const int n0 = (blockIdx.x * wpb + wave) * (16 * NT);
    const int m0 = blockIdx.y * 16;
    if (n0 >= Nc || m0 >= M) return;          // wave-uniform: EXEC stays all-1s

    const int z  = blockIdx.z;
    const int bb = z / Hdim;
    const int hh = z % Hdim;
    const long Aoff = bb * sAb + hh * sAh;
    const long Boff = bb * sBb + hh * sBh;
    const long Coff = bb * sCb + hh * sCh;

    // A fragment (16x32 bf16): lane%16 = row; lane<16 -> K 0..15,
    // lane>=16 -> K 16..31; VGPR pairs hold consecutive K.
    const int rowA  = m0 + (lane & 15);
    const int colB  = lane & 15;
    const int kAoff = (lane < 16) ? 0 : 16;
    const int kBoff = (lane < 16) ? 0 : 16;
    const TA* Abase = A + Aoff + (long)rowA * lda + kAoff;

    v8f acc[NT] = {};
    for (int kt = 0; kt < K; kt += 32) {
        v16bf afrag;
        load8(Abase + kt,      afrag, 0);
        load8(Abase + kt + 8,  afrag, 8);
        if (kt + 32 < K)
            __builtin_prefetch((const void*)(Abase + kt + 32), 0, 1);

        #pragma unroll
        for (int nt = 0; nt < NT; ++nt) {
            // B fragment (32x16 bf16): lane%16 = column; lane-half selects
            // K 0-15 vs 16-31; VGPR pairs hold consecutive K.
            v16bf bfrag;
            const int bn = n0 + nt * 16 + colB;
            if constexpr (BT) {
                const TB* Bp = B + Boff + (long)bn * ldb + kt + kBoff;
                load8(Bp,     bfrag, 0);
                load8(Bp + 8, bfrag, 8);
            } else {
                #pragma unroll
                for (int e = 0; e < 16; ++e)
                    bfrag[e] = to_bf16(B[Boff + (long)(kt + kBoff + e) * ldb + bn]);
            }
            acc[nt] = __builtin_amdgcn_wmma_f32_16x16x32_bf16(
                /*neg_a=*/false, afrag, /*neg_b=*/false, bfrag,
                /*c_mod=*/(short)0, acc[nt], /*reuse_a=*/false, /*reuse_b=*/false);
        }
    }

    // C/D layout: VGPR r -> row (lane<16 ? r : r+8), lane%16 = column.
    const int rbase = (lane < 16) ? 0 : 8;
    #pragma unroll
    for (int nt = 0; nt < NT; ++nt) {
        const int col = n0 + nt * 16 + colB;
        float bias = 0.f;
        if constexpr (HAS_BIAS) bias = rowBias[col];
        #pragma unroll
        for (int r = 0; r < 8; ++r) {
            const int row = m0 + rbase + r;
            const long idx = Coff + (long)row * ldc + col;
            float v = acc[nt][r] * alpha + bias;
            if constexpr (HAS_ADD) v += addFull[idx];
            if constexpr (OUT_F) outF[idx] = v;
            if constexpr (OUT_B) outB[idx] = (__bf16)v;
        }
    }
}

// ---------------------------------------------------------------------------
// Kernel 3: row softmax over last dim (512) of S, writes bf16 probabilities.
// One wave32 per row, 16 elements per lane, shuffle reductions.
// ---------------------------------------------------------------------------
__global__ __launch_bounds__(256)
void softmax_rows_kernel(const float* __restrict__ S, __bf16* __restrict__ P)
{
    const long row  = (long)blockIdx.x * 8 + (threadIdx.x >> 5);
    const int  lane = threadIdx.x & 31;
    const float* r = S + row * Nn;

    float vals[16];
    float mx = -3.0e38f;
    #pragma unroll
    for (int i = 0; i < 16; ++i) {
        vals[i] = r[lane + i * 32];
        mx = fmaxf(mx, vals[i]);
    }
    #pragma unroll
    for (int o = 16; o >= 1; o >>= 1) mx = fmaxf(mx, __shfl_xor(mx, o, 32));

    float sum = 0.f;
    #pragma unroll
    for (int i = 0; i < 16; ++i) {
        vals[i] = __expf(vals[i] - mx);
        sum += vals[i];
    }
    #pragma unroll
    for (int o = 16; o >= 1; o >>= 1) sum += __shfl_xor(sum, o, 32);

    const float inv = 1.f / sum;
    __bf16* pr = P + row * Nn;
    #pragma unroll
    for (int i = 0; i < 16; ++i)
        pr[lane + i * 32] = (__bf16)(vals[i] * inv);
}

// ---------------------------------------------------------------------------
// Host-side orchestration. Workspace layout (bytes):
//   wgS : Bc*Hh*Nn*Nn*4 = 67,108,864   (geo log-bias, reused in place for S)
//   PB  : Bc*Hh*Nn*Nn*2 = 33,554,432   (softmax probs, bf16)
//   Qb/Kb/Vb/Ob : 4 * Bc*Nn*Dd*2 = 16,777,216
// total ~117.4 MB (fits inside the 192 MB L2).
// ---------------------------------------------------------------------------
extern "C" void kernel_launch(void* const* d_in, const int* in_sizes, int n_in,
                              void* d_out, int out_size, void* d_ws, size_t ws_size,
                              hipStream_t stream)
{
    const float* input_query = (const float*)d_in[0];
    const float* input_key   = (const float*)d_in[1];
    const float* input_value = (const float*)d_in[2];
    const float* input_box   = (const float*)d_in[3];
    const float* Wq = (const float*)d_in[4];
    const float* bq = (const float*)d_in[5];
    const float* Wk = (const float*)d_in[6];
    const float* bk = (const float*)d_in[7];
    const float* Wv = (const float*)d_in[8];
    const float* bv = (const float*)d_in[9];
    const float* Wo = (const float*)d_in[10];
    const float* bo = (const float*)d_in[11];
    const float* WGw = (const float*)d_in[12];
    const float* WGb = (const float*)d_in[13];
    float* out = (float*)d_out;

    char* ws = (char*)d_ws;
    float* wgS = (float*)ws;  ws += (size_t)Bc * Hh * Nn * Nn * sizeof(float);
    __bf16* PB = (__bf16*)ws; ws += (size_t)Bc * Hh * Nn * Nn * sizeof(__bf16);
    __bf16* Qb = (__bf16*)ws; ws += (size_t)Bc * Nn * Dd * sizeof(__bf16);
    __bf16* Kb = (__bf16*)ws; ws += (size_t)Bc * Nn * Dd * sizeof(__bf16);
    __bf16* Vb = (__bf16*)ws; ws += (size_t)Bc * Nn * Dd * sizeof(__bf16);
    __bf16* Ob = (__bf16*)ws; ws += (size_t)Bc * Nn * Dd * sizeof(__bf16);

    const dim3 blk128(128, 1, 1);
    const dim3 blk32(32, 1, 1);

    // 1) geometric log-bias (B,H,N,N)
    geo_bias_kernel<<<(Bc * Nn * Nn) / 256, 256, 0, stream>>>(input_box, WGw, WGb, wgS);

    // 2) QKV projections: (B*N,D) x (D,D)^T + bias -> bf16
    //    grid: 512 cols / (4 waves * 64 cols) = 2; 4096 rows / 16 = 256
    const dim3 gProj(Dd / (4 * 64), (Bc * Nn) / 16, 1);
    wmma_gemm_kernel<true, float, float, true, false, false, true, 4>
        <<<gProj, blk128, 0, stream>>>(
        input_query, Wq, bq, nullptr, 1.f, nullptr, Qb,
        Bc * Nn, Dd, Dd, Dd, Dd, Dd, 1, 0, 0, 0, 0, 0, 0);
    wmma_gemm_kernel<true, float, float, true, false, false, true, 4>
        <<<gProj, blk128, 0, stream>>>(
        input_key, Wk, bk, nullptr, 1.f, nullptr, Kb,
        Bc * Nn, Dd, Dd, Dd, Dd, Dd, 1, 0, 0, 0, 0, 0, 0);
    wmma_gemm_kernel<true, float, float, true, false, false, true, 4>
        <<<gProj, blk128, 0, stream>>>(
        input_value, Wv, bv, nullptr, 1.f, nullptr, Vb,
        Bc * Nn, Dd, Dd, Dd, Dd, Dd, 1, 0, 0, 0, 0, 0, 0);

    // 3) scores: S[b,h] = wgLog[b,h] + (Q_h K_h^T) / sqrt(DK), in place over wgS
    const dim3 gScore(Nn / (4 * 64), Nn / 16, Bc * Hh);
    wmma_gemm_kernel<true, __bf16, __bf16, false, true, true, false, 4>
        <<<gScore, blk128, 0, stream>>>(
        Qb, Kb, nullptr, wgS, 0.125f, wgS, nullptr,
        Nn, Nn, Dk, Dd, Dd, Nn, Hh,
        (long)Nn * Dd, (long)Dk, (long)Nn * Dd, (long)Dk,
        (long)Hh * Nn * Nn, (long)Nn * Nn);

    // 4) softmax over last dim -> bf16 probabilities
    softmax_rows_kernel<<<(Bc * Hh * Nn) / 8, 256, 0, stream>>>(wgS, PB);

    // 5) context: O[b,:,h*64..] = P[b,h] @ V_h (B non-transposed).
    //    One wave covers the full 64-wide output strip -> 1-wave blocks.
    const dim3 gAV(1, Nn / 16, Bc * Hh);
    wmma_gemm_kernel<false, __bf16, __bf16, false, false, false, true, 4>
        <<<gAV, blk32, 0, stream>>>(
        PB, Vb, nullptr, nullptr, 1.f, nullptr, Ob,
        Nn, Dk, Nn, Nn, Dd, Dd, Hh,
        (long)Hh * Nn * Nn, (long)Nn * Nn, (long)Nn * Dd, (long)Dk,
        (long)Nn * Dd, (long)Dk);

    // 6) output projection: (B*N,D) x Wo^T + bo -> fp32 d_out
    wmma_gemm_kernel<true, __bf16, float, true, false, true, false, 4>
        <<<gProj, blk128, 0, stream>>>(
        Ob, Wo, bo, nullptr, 1.f, out, nullptr,
        Bc * Nn, Dd, Dd, Dd, Dd, Dd, 1, 0, 0, 0, 0, 0, 0);
}
